// HungarianMatcher_9560597201607
// MI455X (gfx1250) — compile-verified, matrix operand order checked by reading, not run on hardware
//
#include <hip/hip_runtime.h>
#include <hip/hip_bf16.h>

// ---------------------------------------------------------------------------
// DETR-style Hungarian matcher for MI455X (gfx1250, wave32, WMMA + TDM).
//  B=64, Q=900, T=50, C=92 (91 kept classes + background dropped).
//  out = [cost f32 [B,Q,T]] ++ [indices [B,T,2] written as float values]
// ---------------------------------------------------------------------------

#define BATCH     64
#define NQ        900
#define NT        50
#define NCLS      92          // logits width
#define KPAD      96          // 92 padded to multiple of 32 for WMMA K
#define QTILE     128         // queries per block (8 waves x 16 rows)
#define QTILES    ((NQ + QTILE - 1) / QTILE)   // 8
#define KSTEPS    (KPAD / 32) // 3
#define NTILES    4           // 50 targets padded to 4 N-tiles of 16

typedef __attribute__((ext_vector_type(16))) _Float16 v16h;
typedef __attribute__((ext_vector_type(8)))  _Float16 v8h;
typedef __attribute__((ext_vector_type(2)))  _Float16 v2h;
typedef __attribute__((ext_vector_type(8)))  float    v8f;
typedef __attribute__((ext_vector_type(4)))  unsigned int v4u;
typedef __attribute__((ext_vector_type(8)))  int      v8i;
typedef __attribute__((ext_vector_type(4)))  int      v4i;

// ---------------------------------------------------------------------------
// Tensor Data Mover: 1-row tile of `nelem` 4-byte elements, global -> LDS.
// D# layout per CDNA5 ISA §8 (group0: count/lds_addr/global_addr/type=2,
// group1: data_size=4B, tensor_dim0=tile_dim0=nelem, dim1=1, stride0=nelem).
// ---------------------------------------------------------------------------
static __device__ inline void tdm_load_row_f32(unsigned int lds_off,
                                               const void* gptr,
                                               unsigned int nelem) {
    unsigned long long ga = (unsigned long long)gptr;
    v4u g0;
    g0[0] = 1u;                                       // count=1, user mode
    g0[1] = lds_off;                                  // LDS byte address
    g0[2] = (unsigned int)(ga & 0xFFFFFFFFu);         // global_addr[31:0]
    g0[3] = (unsigned int)((ga >> 32) & 0x1FFFFFFu)   // global_addr[56:32]
          | (2u << 30);                               // type = 2 ("image")
    v8i g1;
    g1[0] = (int)(2u << 16);                          // data_size = 4 bytes
    g1[1] = (int)((nelem & 0xFFFFu) << 16);           // tensor_dim0[15:0]
    g1[2] = (int)(((nelem >> 16) & 0xFFFFu) | (1u << 16)); // dim0 hi | tensor_dim1=1
    g1[3] = (int)((nelem & 0xFFFFu) << 16);           // tensor_dim1 hi=0 | tile_dim0
    g1[4] = 1;                                        // tile_dim1=1, tile_dim2=0
    g1[5] = (int)nelem;                               // tensor_dim0_stride[31:0]
    g1[6] = 0;
    g1[7] = 0;
    v4i z4 = {0, 0, 0, 0};
#if __clang_major__ >= 23
    v8i z8 = {0, 0, 0, 0, 0, 0, 0, 0};
    __builtin_amdgcn_tensor_load_to_lds(g0, g1, z4, z4, z8, 0);
#else
    __builtin_amdgcn_tensor_load_to_lds(g0, g1, z4, z4, 0);
#endif
}

// ---------------------------------------------------------------------------
// Kernel 1: fused cost matrix. One block = one batch x 128-query tile.
// cost_class computed as  -(softmax(logits) @ OneHot)  on the WMMA pipes.
// Fragment convention (16-bit A 16x32, mirrored for B):
//   lane <16 : K {0..7, 16..23} in halves j 0..15, row/col = lane
//   lane>=16 : K {8..15, 24..31},               row/col = lane-16
// ---------------------------------------------------------------------------
__global__ __launch_bounds__(256)
void cost_matrix_kernel(const float* __restrict__ pred_logits,  // [B,Q,92]
                        const float* __restrict__ pred_boxes,   // [B,Q,4]
                        const int*   __restrict__ tgt_labels,   // [B,T]
                        const float* __restrict__ tgt_boxes,    // [B,T,4]
                        float* __restrict__ cost_out)           // [B,Q,T]
{
    __shared__ _Float16 sh_probA[QTILE][KPAD] __attribute__((aligned(16)));   // 24 KB
    __shared__ _Float16 sh_bfrag[KSTEPS][NTILES][32][16] __attribute__((aligned(16))); // 12 KB
    __shared__ float    sh_pred[QTILE][9];       //  4.5 KB cx,cy,w,h,x0,y0,x1,y1,area
    __shared__ float    sh_tgtx[NT][9];          //  1.8 KB same layout for targets
    __shared__ float    sh_traw[NT * 4] __attribute__((aligned(16)));  // TDM: boxes
    __shared__ int      sh_tlab[NT]     __attribute__((aligned(16)));  // TDM: labels

    const int tid   = threadIdx.x;
    const int wave  = tid >> 5;
    const int lane  = tid & 31;
    const int half  = lane >> 4;           // K-half selector for A/B fragments
    const int rowm  = lane & 15;           // row (A) / col (B,C) within tile
    const int b     = blockIdx.x / QTILES;
    const int qbase = (blockIdx.x % QTILES) * QTILE;

    // ---- Phase 0: TDM-stage this batch's targets into LDS (wave 0) --------
    if (wave == 0) {
        unsigned int off_boxes = (unsigned int)(unsigned long long)(void*)sh_traw;
        unsigned int off_lab   = (unsigned int)(unsigned long long)(void*)sh_tlab;
        tdm_load_row_f32(off_boxes, tgt_boxes + (size_t)b * NT * 4, NT * 4);
        tdm_load_row_f32(off_lab,   tgt_labels + (size_t)b * NT,    NT);
        __builtin_amdgcn_s_wait_tensorcnt(0);
    }
    __syncthreads();

    // ---- Phase 1a: zero the pre-swizzled one-hot B fragments --------------
    {
        unsigned int* p = (unsigned int*)&sh_bfrag[0][0][0][0];
        #pragma unroll
        for (int i = tid; i < (KSTEPS * NTILES * 32 * 16) / 2; i += 256) p[i] = 0u;
    }
    __syncthreads();

    // ---- Phase 1b: one-hot (fragment-layout) + target geometry ------------
    if (tid < NT) {
        const int t   = tid;
        const int lbl = sh_tlab[t];                 // in [0, 91)
        const int kk  = lbl >> 5;                   // K-step
        const int kl  = lbl & 31;                   // k within step
        const int blk = kl >> 3;                    // 8-block 0..3
        const int ln  = (blk & 1) ? (16 + (t & 15)) : (t & 15);
        const int j   = (blk & 2) ? (8 + (kl & 7)) : (kl & 7);
        sh_bfrag[kk][t >> 4][ln][j] = (_Float16)1.0f;

        float cx = sh_traw[t * 4 + 0], cy = sh_traw[t * 4 + 1];
        float w  = sh_traw[t * 4 + 2], h  = sh_traw[t * 4 + 3];
        sh_tgtx[t][0] = cx; sh_tgtx[t][1] = cy; sh_tgtx[t][2] = w; sh_tgtx[t][3] = h;
        sh_tgtx[t][4] = cx - 0.5f * w; sh_tgtx[t][5] = cy - 0.5f * h;
        sh_tgtx[t][6] = cx + 0.5f * w; sh_tgtx[t][7] = cy + 0.5f * h;
        sh_tgtx[t][8] = w * h;
    }

    // ---- Phase 1c: pred geometry + softmax -> f16 A rows ------------------
    for (int ql = tid; ql < QTILE; ql += 256) {
        const int q = qbase + ql;
        v2h* Pr2 = (v2h*)&sh_probA[ql][0];
        if (q < NQ) {
            const float* pb = pred_boxes + ((size_t)b * NQ + q) * 4;
            float cx = pb[0], cy = pb[1], w = pb[2], h = pb[3];
            sh_pred[ql][0] = cx; sh_pred[ql][1] = cy;
            sh_pred[ql][2] = w;  sh_pred[ql][3] = h;
            sh_pred[ql][4] = cx - 0.5f * w; sh_pred[ql][5] = cy - 0.5f * h;
            sh_pred[ql][6] = cx + 0.5f * w; sh_pred[ql][7] = cy + 0.5f * h;
            sh_pred[ql][8] = w * h;

            const float4* L4 = (const float4*)(pred_logits + ((size_t)b * NQ + q) * NCLS);
            __builtin_prefetch(L4, 0, 0);
            float m = -3.402823466e38f;
            #pragma unroll 4
            for (int c4 = 0; c4 < NCLS / 4; ++c4) {
                float4 v = L4[c4];
                m = fmaxf(m, fmaxf(fmaxf(v.x, v.y), fmaxf(v.z, v.w)));
            }
            float s = 0.0f;
            #pragma unroll 4
            for (int c4 = 0; c4 < NCLS / 4; ++c4) {
                float4 v = L4[c4];
                s += (__expf(v.x - m) + __expf(v.y - m)) +
                     (__expf(v.z - m) + __expf(v.w - m));
            }
            const float inv = 1.0f / s;
            #pragma unroll 4
            for (int c4 = 0; c4 < NCLS / 4; ++c4) {
                float4 v = L4[c4];
                v2h p01, p23;
                p01[0] = (_Float16)(__expf(v.x - m) * inv);
                p01[1] = (_Float16)(__expf(v.y - m) * inv);
                p23[0] = (_Float16)(__expf(v.z - m) * inv);
                p23[1] = (_Float16)(__expf(v.w - m) * inv);
                Pr2[2 * c4]     = p01;
                Pr2[2 * c4 + 1] = p23;
            }
            Pr2[46] = (v2h){(_Float16)0.0f, (_Float16)0.0f};   // pad 92..95
            Pr2[47] = (v2h){(_Float16)0.0f, (_Float16)0.0f};
        } else {
            #pragma unroll
            for (int i = 0; i < KPAD / 2; ++i)
                Pr2[i] = (v2h){(_Float16)0.0f, (_Float16)0.0f};
            #pragma unroll
            for (int j = 0; j < 9; ++j) sh_pred[ql][j] = 0.0f;
        }
    }
    __syncthreads();

    // ---- Phase 2: class-cost GEMM on WMMA (b128 fragment loads) -----------
    v8f acc[NTILES];
    #pragma unroll
    for (int n = 0; n < NTILES; ++n)
        acc[n] = (v8f){0.f, 0.f, 0.f, 0.f, 0.f, 0.f, 0.f, 0.f};

    const _Float16* Arow = &sh_probA[16 * wave + rowm][0];
    #pragma unroll
    for (int kk = 0; kk < KSTEPS; ++kk) {
        const int k0 = kk * 32;
        const v8h a0 = *(const v8h*)(Arow + k0 + half * 8);
        const v8h a1 = *(const v8h*)(Arow + k0 + 16 + half * 8);
        const v16h a = __builtin_shufflevector(a0, a1,
                        0, 1, 2, 3, 4, 5, 6, 7, 8, 9, 10, 11, 12, 13, 14, 15);
        #pragma unroll
        for (int n = 0; n < NTILES; ++n) {
            const v8h* bp = (const v8h*)&sh_bfrag[kk][n][lane][0];
            const v8h b0 = bp[0];
            const v8h b1 = bp[1];
            const v16h bf = __builtin_shufflevector(b0, b1,
                        0, 1, 2, 3, 4, 5, 6, 7, 8, 9, 10, 11, 12, 13, 14, 15);
            acc[n] = __builtin_amdgcn_wmma_f32_16x16x32_f16(
                         false, a, false, bf, (short)0, acc[n], false, false);
        }
    }

    // ---- Phase 3: fused epilogue (L1 bbox + GIoU + class) -----------------
    // C/D layout: lane<16 -> N=lane, M=i ; lane>=16 -> N=lane-16, M=8+i.
    #pragma unroll
    for (int n = 0; n < NTILES; ++n) {
        const int t = 16 * n + rowm;
        if (t >= NT) continue;
        const float tcx = sh_tgtx[t][0], tcy = sh_tgtx[t][1];
        const float tw  = sh_tgtx[t][2], th  = sh_tgtx[t][3];
        const float bx0 = sh_tgtx[t][4], by0 = sh_tgtx[t][5];
        const float bx1 = sh_tgtx[t][6], by1 = sh_tgtx[t][7];
        const float areab = sh_tgtx[t][8];
        #pragma unroll
        for (int i = 0; i < 8; ++i) {
            const int ql = 16 * wave + half * 8 + i;
            const int q  = qbase + ql;
            if (q >= NQ) continue;
            const float prob = acc[n][i];            // softmax prob @ label_t
            const float cx = sh_pred[ql][0], cy = sh_pred[ql][1];
            const float w  = sh_pred[ql][2], h  = sh_pred[ql][3];
            const float ax0 = sh_pred[ql][4], ay0 = sh_pred[ql][5];
            const float ax1 = sh_pred[ql][6], ay1 = sh_pred[ql][7];
            const float areaa = sh_pred[ql][8];
            const float bbox = fabsf(cx - tcx) + fabsf(cy - tcy) +
                               fabsf(w - tw)   + fabsf(h - th);
            const float iw = fmaxf(fminf(ax1, bx1) - fmaxf(ax0, bx0), 0.0f);
            const float ih = fmaxf(fminf(ay1, by1) - fmaxf(ay0, by0), 0.0f);
            const float inter = iw * ih;
            const float uni   = areaa + areab - inter;
            const float iou   = inter / uni;
            const float ew = fmaxf(fmaxf(ax1, bx1) - fminf(ax0, bx0), 0.0f);
            const float eh = fmaxf(fmaxf(ay1, by1) - fminf(ay0, by0), 0.0f);
            const float areae = ew * eh;
            const float giou  = iou - (areae - uni) / areae;
            const float cost  = 5.0f * bbox - prob - 2.0f * giou;
            cost_out[((size_t)b * NQ + q) * NT + t] = cost;
        }
    }
}

// ---------------------------------------------------------------------------
// Kernel 2: greedy assignment. One block per batch, 50 masked argmin rounds,
// row-major-first tie-break (matches jnp.argmin). Wave32 shuffle butterflies
// keep the serial critical path to 2 barriers per round.
// ---------------------------------------------------------------------------
#define GREEDY_THREADS 512
#define GREEDY_WAVES   (GREEDY_THREADS / 32)

__global__ __launch_bounds__(GREEDY_THREADS)
void greedy_assign_kernel(const float* __restrict__ cost,   // [B,Q,T]
                          float* __restrict__ idx_out)      // [B,T,2] as floats
{
    __shared__ float         wvals[GREEDY_WAVES];
    __shared__ int           widx[GREEDY_WAVES];
    __shared__ unsigned char row_used[NQ];
    __shared__ int           col_used[NT];

    const int b    = blockIdx.x;
    const int tid  = threadIdx.x;
    const int lane = tid & 31;
    const int wv   = tid >> 5;
    const float* C = cost + (size_t)b * NQ * NT;

    for (int i = tid; i < NQ; i += GREEDY_THREADS) row_used[i] = 0;
    if (tid < NT) col_used[tid] = 0;
    __syncthreads();

    for (int step = 0; step < NT; ++step) {
        float best = 3.402823466e38f;
        int   bidx = 0x7FFFFFFF;
        for (int i = tid; i < NQ * NT; i += GREEDY_THREADS) {
            const int r = i / NT;
            const int c = i - r * NT;
            if (row_used[r] | col_used[c]) continue;
            const float v = C[i];
            if (v < best || (v == best && i < bidx)) { best = v; bidx = i; }
        }
        // wave32 butterfly min-reduce (value, then first row-major index)
        #pragma unroll
        for (int m = 16; m > 0; m >>= 1) {
            const float ov = __shfl_xor(best, m, 32);
            const int   oi = __shfl_xor(bidx, m, 32);
            if (ov < best || (ov == best && oi < bidx)) { best = ov; bidx = oi; }
        }
        if (lane == 0) { wvals[wv] = best; widx[wv] = bidx; }
        __syncthreads();
        if (tid < 32) {
            float v = (lane < GREEDY_WAVES) ? wvals[lane] : 3.402823466e38f;
            int   ix = (lane < GREEDY_WAVES) ? widx[lane] : 0x7FFFFFFF;
            #pragma unroll
            for (int m = 16; m > 0; m >>= 1) {
                const float ov = __shfl_xor(v, m, 32);
                const int   oi = __shfl_xor(ix, m, 32);
                if (ov < v || (ov == v && oi < ix)) { v = ov; ix = oi; }
            }
            if (lane == 0) {
                const int r = ix / NT;
                const int c = ix - r * NT;
                row_used[r] = 1;
                col_used[c] = 1;
                idx_out[((size_t)b * NT + step) * 2 + 0] = (float)r;
                idx_out[((size_t)b * NT + step) * 2 + 1] = (float)c;
            }
        }
        __syncthreads();
    }
}

// ---------------------------------------------------------------------------
extern "C" void kernel_launch(void* const* d_in, const int* in_sizes, int n_in,
                              void* d_out, int out_size, void* d_ws, size_t ws_size,
                              hipStream_t stream) {
    (void)in_sizes; (void)n_in; (void)d_ws; (void)ws_size; (void)out_size;
    const float* pred_logits = (const float*)d_in[0];   // [64,900,92] f32
    const float* pred_boxes  = (const float*)d_in[1];   // [64,900,4]  f32
    const int*   tgt_labels  = (const int*)  d_in[2];   // [64,50]     i32
    const float* tgt_boxes   = (const float*)d_in[3];   // [64,50,4]   f32

    float* cost_out = (float*)d_out;                      // 64*900*50 floats
    float* idx_out  = cost_out + (size_t)BATCH * NQ * NT; // then 64*50*2

    cost_matrix_kernel<<<BATCH * QTILES, 256, 0, stream>>>(
        pred_logits, pred_boxes, tgt_labels, tgt_boxes, cost_out);
    greedy_assign_kernel<<<BATCH, GREEDY_THREADS, 0, stream>>>(cost_out, idx_out);
}
